// APPNP_82935818486074
// MI455X (gfx1250) — matrix operand order repaired and checked by reading, not compiled, not run on hardware
//
#include <hip/hip_runtime.h>

#define FIN   512
#define HDIM  64
#define CDIM  40
#define CPAD  48
#define ALPHA 0.1f

typedef __attribute__((ext_vector_type(16))) __bf16 v16bf;
typedef __attribute__((ext_vector_type(8)))  __bf16 v8bf;
typedef __attribute__((ext_vector_type(8)))  float  v8f;
typedef __attribute__((ext_vector_type(4)))  float  v4f;

#define W1_STRIDE 520   // bf16 elems per row of W1^T tile (pad vs 512 to break bank conflicts)
#define W2_STRIDE 72
#define H_STRIDE  72

static __device__ __forceinline__ v16bf cat8(v8bf lo, v8bf hi) {
    return __builtin_shufflevector(lo, hi, 0,1,2,3,4,5,6,7,8,9,10,11,12,13,14,15);
}

// ---------------------------------------------------------------------------
// Fused MLP: h0 = relu(x @ W1 + b1) @ W2 + b2   (bf16 WMMA, f32 accumulate)
// 128 threads = 4 waves; each wave owns a 16-row tile (block covers 64 nodes).
// ---------------------------------------------------------------------------
__launch_bounds__(128)
__global__ void appnp_mlp_kernel(const float* __restrict__ x,
                                 const float* __restrict__ W1,
                                 const float* __restrict__ b1,
                                 const float* __restrict__ W2,
                                 const float* __restrict__ b2,
                                 float* __restrict__ h0,
                                 int N) {
    __shared__ __bf16 sW1t[HDIM * W1_STRIDE];     // W1 transposed: [n][k]
    __shared__ __bf16 sW2t[CPAD * W2_STRIDE];     // W2 transposed, rows 40..47 zero
    __shared__ __bf16 sH[4 * 16 * H_STRIDE];      // per-wave hidden tiles

    const int tid  = threadIdx.x;
    const int wave = tid >> 5;
    const int lane = tid & 31;
    const int mrow = lane & 15;
    const int kh   = lane >> 4;

    // Stage W1^T (coalesced read of W1[k][n], f32 -> bf16)
    for (int i = tid; i < FIN * HDIM; i += blockDim.x) {
        int k = i >> 6, n = i & 63;
        sW1t[n * W1_STRIDE + k] = (__bf16)W1[i];
    }
    // Stage W2^T and zero the padded rows
    for (int i = tid; i < HDIM * CDIM; i += blockDim.x) {
        int k = i / CDIM, c = i - k * CDIM;
        sW2t[c * W2_STRIDE + k] = (__bf16)W2[i];
    }
    for (int i = tid; i < (CPAD - CDIM) * HDIM; i += blockDim.x) {
        int c = CDIM + (i >> 6), k = i & 63;
        sW2t[c * W2_STRIDE + k] = (__bf16)0.0f;
    }
    __syncthreads();

    const int rowBase = blockIdx.x * 64 + wave * 16;
    int srcRow = rowBase + mrow;
    if (srcRow >= N) srcRow = N - 1;               // clamp (stores are guarded)
    const float* xr = x + (size_t)srcRow * FIN;

    // ---------------- GEMM1: [16x512] x [512x64] ----------------
    v8f acc[4] = {v8f{}, v8f{}, v8f{}, v8f{}};
    for (int k0 = 0; k0 < FIN; k0 += 32) {
        // A fragment: lanes 0-15 hold K {0..7,16..23}, lanes 16-31 hold K {8..15,24..31}
        v16bf afr;
        {
            const float* p0 = xr + k0 + kh * 8;
            const float* p1 = xr + k0 + 16 + kh * 8;
            v4f a0 = *(const v4f*)(p0);
            v4f a1 = *(const v4f*)(p0 + 4);
            v4f a2 = *(const v4f*)(p1);
            v4f a3 = *(const v4f*)(p1 + 4);
            #pragma unroll
            for (int j = 0; j < 4; ++j) {
                afr[j]      = (__bf16)a0[j];
                afr[4 + j]  = (__bf16)a1[j];
                afr[8 + j]  = (__bf16)a2[j];
                afr[12 + j] = (__bf16)a3[j];
            }
        }
        #pragma unroll
        for (int t = 0; t < 4; ++t) {
            const __bf16* wp = &sW1t[(t * 16 + mrow) * W1_STRIDE + k0 + kh * 8];
            v16bf bfr = cat8(*(const v8bf*)wp, *(const v8bf*)(wp + 16));
            acc[t] = __builtin_amdgcn_wmma_f32_16x16x32_bf16(
                false, afr, false, bfr, (short)0, acc[t], false, false);
        }
    }

    // relu + b1, write hidden tile to wave-private LDS region as bf16
    __bf16* hrow = &sH[wave * 16 * H_STRIDE];
    #pragma unroll
    for (int t = 0; t < 4; ++t) {
        int cc = t * 16 + mrow;
        float bias = b1[cc];
        #pragma unroll
        for (int r = 0; r < 8; ++r) {
            int m = kh * 8 + r;                    // D layout: lane group selects row half
            float v = acc[t][r] + bias;
            hrow[m * H_STRIDE + cc] = (__bf16)(v > 0.0f ? v : 0.0f);
        }
    }
    __syncthreads();

    // ---------------- GEMM2: [16x64] x [64x48(pad of 40)] ----------------
    v8f oacc[3] = {v8f{}, v8f{}, v8f{}};
    const __bf16* hb = &sH[wave * 16 * H_STRIDE + mrow * H_STRIDE];
    #pragma unroll
    for (int k0 = 0; k0 < HDIM; k0 += 32) {
        v16bf afr = cat8(*(const v8bf*)(hb + k0 + kh * 8),
                         *(const v8bf*)(hb + k0 + 16 + kh * 8));
        #pragma unroll
        for (int t = 0; t < 3; ++t) {
            const __bf16* wp = &sW2t[(t * 16 + mrow) * W2_STRIDE + k0 + kh * 8];
            v16bf bfr = cat8(*(const v8bf*)wp, *(const v8bf*)(wp + 16));
            oacc[t] = __builtin_amdgcn_wmma_f32_16x16x32_bf16(
                false, afr, false, bfr, (short)0, oacc[t], false, false);
        }
    }

    // + b2, store [16 x 40] tile
    #pragma unroll
    for (int t = 0; t < 3; ++t) {
        int cc = t * 16 + mrow;
        float bias = (cc < CDIM) ? b2[cc] : 0.0f;
        #pragma unroll
        for (int r = 0; r < 8; ++r) {
            int gr = rowBase + kh * 8 + r;
            if (cc < CDIM && gr < N)
                h0[(size_t)gr * CDIM + cc] = oacc[t][r] + bias;
        }
    }
}

// ---------------------------------------------------------------------------
// Graph-norm precompute
// ---------------------------------------------------------------------------
__global__ void appnp_deg_kernel(const int* __restrict__ col, float* __restrict__ deg, int E) {
    int e = blockIdx.x * blockDim.x + threadIdx.x;
    if (e < E) unsafeAtomicAdd(&deg[col[e]], 1.0f);
}

__global__ void appnp_dinv_kernel(const float* __restrict__ deg, float* __restrict__ dinv, int N) {
    int n = blockIdx.x * blockDim.x + threadIdx.x;
    if (n < N) dinv[n] = __frsqrt_rn(deg[n] + 1.0f);   // +1 self loop => deg > 0 always
}

__global__ void appnp_norm_kernel(const int* __restrict__ row, const int* __restrict__ col,
                                  const float* __restrict__ dinv, float* __restrict__ nrm, int E) {
    int e = blockIdx.x * blockDim.x + threadIdx.x;
    if (e < E) nrm[e] = dinv[row[e]] * dinv[col[e]];
}

// ---------------------------------------------------------------------------
// Propagation step:  h_next = ALPHA*x0 + (1-ALPHA)*(self-loop + sum_in msg)
// init kernel writes the alpha-term plus the self-loop contribution, then the
// edge kernel scatter-adds (1-ALPHA)*norm[e]*h_cur[row] with fp32 atomics.
// ---------------------------------------------------------------------------
__global__ void appnp_init_kernel(const float* __restrict__ x0, const float* __restrict__ hc,
                                  const float* __restrict__ dinv, float* __restrict__ hn,
                                  int total /* N*CDIM */) {
    int i = blockIdx.x * blockDim.x + threadIdx.x;
    if (i >= total) return;
    int n = i / CDIM;
    float di = dinv[n];
    hn[i] = ALPHA * x0[i] + (1.0f - ALPHA) * di * di * hc[i];
}

__global__ void appnp_edge_kernel(const int* __restrict__ row, const int* __restrict__ col,
                                  const float* __restrict__ nrm, const float* __restrict__ hc,
                                  float* __restrict__ hn, int E) {
    int t = blockIdx.x * blockDim.x + threadIdx.x;
    int e = t >> 3;
    if (e >= E) return;
    int c0 = (t & 7) * 5;                          // 8 lanes/edge x 5 channels = 40
    float w = (1.0f - ALPHA) * nrm[e];
    const float* src = hc + (size_t)row[e] * CDIM + c0;
    float* dst = hn + (size_t)col[e] * CDIM + c0;
    #pragma unroll
    for (int i = 0; i < 5; ++i)
        unsafeAtomicAdd(dst + i, w * src[i]);
}

// ---------------------------------------------------------------------------
// log_softmax over C=40, one thread per node (row fits in VGPRs)
// ---------------------------------------------------------------------------
__global__ void appnp_logsm_kernel(const float* __restrict__ h, float* __restrict__ out, int N) {
    int n = blockIdx.x * blockDim.x + threadIdx.x;
    if (n >= N) return;
    const float* r = h + (size_t)n * CDIM;
    float v[CDIM];
    float m = -3.0e38f;
    #pragma unroll
    for (int c = 0; c < CDIM; ++c) { v[c] = r[c]; m = fmaxf(m, v[c]); }
    float s = 0.0f;
    #pragma unroll
    for (int c = 0; c < CDIM; ++c) { v[c] -= m; s += __expf(v[c]); }
    float ls = __logf(s);
    float* o = out + (size_t)n * CDIM;
    #pragma unroll
    for (int c = 0; c < CDIM; ++c) o[c] = v[c] - ls;
}

// ---------------------------------------------------------------------------
extern "C" void kernel_launch(void* const* d_in, const int* in_sizes, int n_in,
                              void* d_out, int out_size, void* d_ws, size_t ws_size,
                              hipStream_t stream) {
    const float* x  = (const float*)d_in[0];
    const int*   ei = (const int*)d_in[1];
    const float* W1 = (const float*)d_in[2];
    const float* b1 = (const float*)d_in[3];
    const float* W2 = (const float*)d_in[4];
    const float* b2 = (const float*)d_in[5];

    const int N = in_sizes[0] / FIN;
    const int E = in_sizes[1] / 2;
    const int* row = ei;        // edge_index[0] = source
    const int* col = ei + E;    // edge_index[1] = target

    float* ws   = (float*)d_ws;
    float* x0   = ws;  ws += (size_t)N * CDIM;
    float* bufA = ws;  ws += (size_t)N * CDIM;
    float* bufB = ws;  ws += (size_t)N * CDIM;
    float* deg  = ws;  ws += N;
    float* dinv = ws;  ws += N;
    float* nrm  = ws;  // E floats

    hipMemsetAsync(deg, 0, sizeof(float) * (size_t)N, stream);

    appnp_mlp_kernel<<<(N + 63) / 64, 128, 0, stream>>>(x, W1, b1, W2, b2, x0, N);
    appnp_deg_kernel<<<(E + 255) / 256, 256, 0, stream>>>(col, deg, E);
    appnp_dinv_kernel<<<(N + 255) / 256, 256, 0, stream>>>(deg, dinv, N);
    appnp_norm_kernel<<<(E + 255) / 256, 256, 0, stream>>>(row, col, dinv, nrm, E);

    const float* hc = x0;
    float* hn = bufA;
    const int totNC = N * CDIM;
    for (int k = 0; k < 10; ++k) {
        appnp_init_kernel<<<(totNC + 255) / 256, 256, 0, stream>>>(x0, hc, dinv, hn, totNC);
        appnp_edge_kernel<<<((E * 8) + 255) / 256, 256, 0, stream>>>(row, col, nrm, hc, hn, E);
        hc = hn;
        hn = (hn == bufA) ? bufB : bufA;
    }

    appnp_logsm_kernel<<<(N + 127) / 128, 128, 0, stream>>>(hc, (float*)d_out, N);
}
